// MultiHeadAttention_6983616824118
// MI455X (gfx1250) — matrix operand order repaired and checked by reading, not compiled, not run on hardware
//
#include <hip/hip_runtime.h>
#include <hip/hip_bf16.h>

// ---------------------------------------------------------------------------
// MultiHeadAttention for MI455X (gfx1250): all contractions via
// v_wmma_f32_16x16x32_bf16 (wave32, 16x16 tiles, K=32 per instruction).
// K/V tiles staged into LDS via global_load_async_to_lds_b128 (ASYNCcnt).
// ---------------------------------------------------------------------------

typedef __attribute__((ext_vector_type(16))) __bf16   v16bf;
typedef __attribute__((ext_vector_type(8)))  float    v8f;
typedef __attribute__((ext_vector_type(4)))  unsigned int u32x4v;
typedef __attribute__((ext_vector_type(4)))  float    f32x4v;

union FragBF {
    v16bf  v;
    __bf16 e[16];
    u32x4v q[2];
};

static __device__ __forceinline__ v8f wmma_bf16(const v16bf& a, const v16bf& b, v8f c) {
    // D = A(16x32) * B(32x16) + C(16x16), f32 accumulate
    return __builtin_amdgcn_wmma_f32_16x16x32_bf16(false, a, false, b, (short)0, c, false, false);
}

// ---------------------------------------------------------------------------
// Generic GEMM: C[M x N] = A[M x K] * B[K x N]
//   A: f32 (converted to bf16 in-register) or bf16, row-major, leading dim lda
//   B: f32 weights, row-major, leading dim ldb (converted to bf16 per-element)
//   C: bf16 or f32, leading dim ldc
// Block = 128 threads (4 waves). Each wave: one 16x16 C tile.
//   row tile = blockIdx.x*16, col tile = blockIdx.y*64 + wave*16.
// HEADED mode: the N dimension is 8 heads x 64; each 16-col tile lies inside
// one head (head = col0>>6).  B += head*bStride, C += head*cStride, and the
// in-head column is col0&63.  This lets ONE pass over A produce all heads
// (8x less A-traffic than a grid.z=head layout).
// ---------------------------------------------------------------------------
template <bool A_F32, bool OUT_BF16, bool HEADED>
__global__ __launch_bounds__(128)
void gemm_wmma_bf16(const void* __restrict__ Aptr, int lda,
                    const float* __restrict__ Bw, int ldb,
                    void* __restrict__ Cptr, int ldc, int K,
                    long bStride, long cStride)
{
    const int lane = threadIdx.x & 31;
    const int wave = threadIdx.x >> 5;
    const int row0 = blockIdx.x * 16;
    const int col0 = blockIdx.y * 64 + wave * 16;
    const int head = HEADED ? (col0 >> 6) : 0;
    const int colH = HEADED ? (col0 & 63) : col0;

    const float* Bz = Bw + (size_t)head * (size_t)bStride;

    const bool lo  = lane < 16;
    const int rlo  = lane & 15;
    const int kA   = lo ? 0 : 8;    // A fragment K base (16-bit A layout)
    const int kB   = lo ? 0 : 16;   // unused; the B fragment uses kA (same k order as A)
    const int row  = row0 + rlo;

    v8f c = {};
    for (int k0 = 0; k0 < K; k0 += 32) {
        if (k0 + 64 <= K) {
            __builtin_prefetch((const char*)Aptr +
                ((size_t)row * lda + k0 + 64) * (A_F32 ? 4 : 2), 0, 1);
        }
        FragBF a;
        if (A_F32) {
            const float* Af = (const float*)Aptr + (size_t)row * lda + k0 + kA;
            f32x4v x0 = *(const f32x4v*)(Af + 0);
            f32x4v x1 = *(const f32x4v*)(Af + 4);
            f32x4v x2 = *(const f32x4v*)(Af + 16);
            f32x4v x3 = *(const f32x4v*)(Af + 20);
            #pragma unroll
            for (int t = 0; t < 4; ++t) {
                a.e[t]      = (__bf16)x0[t];
                a.e[4 + t]  = (__bf16)x1[t];
                a.e[8 + t]  = (__bf16)x2[t];
                a.e[12 + t] = (__bf16)x3[t];
            }
        } else {
            const __bf16* Ab = (const __bf16*)Aptr + (size_t)row * lda + k0 + kA;
            a.q[0] = *(const u32x4v*)(Ab);
            a.q[1] = *(const u32x4v*)(Ab + 16);
        }
        FragBF b;
        #pragma unroll
        for (int i = 0; i < 16; ++i)
            b.e[i] = (__bf16)Bz[(size_t)(k0 + kA + i + (i & 8)) * ldb + colH + rlo];

        c = wmma_bf16(a.v, b.v, c);
    }

    #pragma unroll
    for (int r = 0; r < 8; ++r) {
        const int rr = row0 + (lo ? r : 8 + r);
        const size_t idx = (size_t)head * (size_t)cStride + (size_t)rr * ldc + colH + rlo;
        if (OUT_BF16) ((__bf16*)Cptr)[idx] = (__bf16)c[r];
        else          ((float*)Cptr)[idx]  = c[r];
    }
}

// ---------------------------------------------------------------------------
// Flash-style attention per (head, batch).
// grid = (NQ/64, B, H), block = 128 (4 waves). Each wave owns 16 query rows.
// K/V streamed in 32-key tiles into LDS via GLOBAL_LOAD_ASYNC_TO_LDS_B128;
// online softmax; all matmuls WMMA.
// Heads output layout: [b*512 + q][h*64 + e] (bf16) so the final projection
// against W_out (h,e,o) is one plain GEMM.
// ---------------------------------------------------------------------------
__global__ __launch_bounds__(128)
void attn_flash_wmma(const __bf16* __restrict__ Qh, const __bf16* __restrict__ Kh,
                     const __bf16* __restrict__ Vh,
                     const unsigned char* __restrict__ mask,  // jax bool: 1 byte/elem
                     __bf16* __restrict__ Heads)
{
    __shared__ alignas(16) __bf16 Qs[64 * 64];     // 64 queries x 64 e
    __shared__ alignas(16) __bf16 Ks[32 * 64];     // 32 keys    x 64 e (4 KB)
    __shared__ alignas(16) __bf16 Vs[32 * 64];     // 32 keys    x 64 e (4 KB)
    __shared__ alignas(16) __bf16 Ps[4][16 * 32];  // per-wave P relayout buffer

    const int b  = blockIdx.y;
    const int h  = blockIdx.z;
    const int q0 = blockIdx.x * 64;

    const size_t slice = ((size_t)h * 32 + b) * (size_t)512 * 64;
    const __bf16* Qsl = Qh + slice + (size_t)q0 * 64;
    const __bf16* Ksl = Kh + slice;
    const __bf16* Vsl = Vh + slice;
    const unsigned char* mb = mask + (size_t)b * 512;

    const int tid  = threadIdx.x;
    const int lane = tid & 31;
    const int wave = tid >> 5;
    const bool lo  = lane < 16;
    const int rlo  = lane & 15;
    const int kA   = lo ? 0 : 8;
    const int kB   = lo ? 0 : 16;

    // LDS byte offsets of the staging tiles (generic-pointer low 32 bits are
    // the wave-relative LDS byte address per the flat-aperture rules).
    const unsigned kldsBase = (unsigned)(size_t)(void*)&Ks[0];
    const unsigned vldsBase = (unsigned)(size_t)(void*)&Vs[0];
    const int aoff0 = tid * 16;          // 128 threads x 16B = 2 KB per op
    const int aoff1 = tid * 16 + 2048;   // second half of the 4 KB tile

    // Cooperative, coalesced load of the 64x64 Q tile.
    {
        const unsigned int* src = (const unsigned int*)Qsl;
        unsigned int* dst = (unsigned int*)Qs;
        for (int i = tid; i < 64 * 64 / 2; i += 128) dst[i] = src[i];
    }
    __syncthreads();

    // Q A-fragments (reused across all key tiles): e-chunks k0 = 0, 32.
    FragBF aq0, aq1;
    {
        const __bf16* qrow = &Qs[(wave * 16 + rlo) * 64];
        aq0.q[0] = *(const u32x4v*)(qrow + 0  + kB);
        aq0.q[1] = *(const u32x4v*)(qrow + 0  + kB + 8);
        aq1.q[0] = *(const u32x4v*)(qrow + 32 + kB);
        aq1.q[1] = *(const u32x4v*)(qrow + 32 + kB + 8);
    }

    v8f o0 = {}, o1 = {}, o2 = {}, o3 = {};   // O accumulator: 16 rows x 64 e
    float m[8], ls[8];
    #pragma unroll
    for (int r = 0; r < 8; ++r) { m[r] = -3.0e38f; ls[r] = 0.0f; }

    for (int kt = 0; kt < 512; kt += 32) {
        __syncthreads();   // previous tile's Ks/Vs consumers done

        // Async DMA of the 32x64 bf16 K and V tiles into LDS (GVS mode:
        // mem = SGPR base + 32-bit VGPR offset; vdst = LDS byte address).
        {
            const __bf16* kg = Ksl + (size_t)kt * 64;
            const __bf16* vg = Vsl + (size_t)kt * 64;
            asm volatile(
                "global_load_async_to_lds_b128 %0, %2, %4\n\t"
                "global_load_async_to_lds_b128 %1, %3, %4"
                :: "v"(kldsBase + aoff0), "v"(kldsBase + aoff1),
                   "v"(aoff0), "v"(aoff1), "s"(kg)
                : "memory");
            asm volatile(
                "global_load_async_to_lds_b128 %0, %2, %4\n\t"
                "global_load_async_to_lds_b128 %1, %3, %4"
                :: "v"(vldsBase + aoff0), "v"(vldsBase + aoff1),
                   "v"(aoff0), "v"(aoff1), "s"(vg)
                : "memory");
            asm volatile("s_wait_asynccnt 0x0" ::: "memory");
        }
        __syncthreads();

        // S = Q * K^T for this wave's 16 rows x 32 keys (two 16x16 C tiles).
        // B fragment of K^T: contraction dim is e, N = key -> contiguous e-run
        // of the LDS K row indexed by this lane's key.
        v8f s0 = {}, s1 = {};
        {
            FragBF bk;
            const __bf16* k0row = &Ks[rlo * 64];
            const __bf16* k1row = &Ks[(16 + rlo) * 64];
            bk.q[0] = *(const u32x4v*)(k0row + 0  + kB);
            bk.q[1] = *(const u32x4v*)(k0row + 0  + kB + 8);
            s0 = wmma_bf16(aq0.v, bk.v, s0);
            bk.q[0] = *(const u32x4v*)(k0row + 32 + kB);
            bk.q[1] = *(const u32x4v*)(k0row + 32 + kB + 8);
            s0 = wmma_bf16(aq1.v, bk.v, s0);
            bk.q[0] = *(const u32x4v*)(k1row + 0  + kB);
            bk.q[1] = *(const u32x4v*)(k1row + 0  + kB + 8);
            s1 = wmma_bf16(aq0.v, bk.v, s1);
            bk.q[0] = *(const u32x4v*)(k1row + 32 + kB);
            bk.q[1] = *(const u32x4v*)(k1row + 32 + kB + 8);
            s1 = wmma_bf16(aq1.v, bk.v, s1);
        }

        // Mask (-1e20 BEFORE norm scaling, as in reference), scale, online softmax.
        const bool mk0 = mb[kt + rlo]      != 0;
        const bool mk1 = mb[kt + 16 + rlo] != 0;
        #pragma unroll
        for (int r = 0; r < 8; ++r) {
            float x0 = mk0 ? -1.0e20f : s0[r];
            float x1 = mk1 ? -1.0e20f : s1[r];
            x0 *= 0.125f; x1 *= 0.125f;                 // 1/sqrt(64)
            float t = fmaxf(x0, x1);
            #pragma unroll
            for (int off = 1; off < 16; off <<= 1)      // reduce across 16 key lanes
                t = fmaxf(t, __shfl_xor(t, off, 32));
            const float mn    = fmaxf(m[r], t);
            const float alpha = __expf(m[r] - mn);
            const float p0    = __expf(x0 - mn);
            const float p1    = __expf(x1 - mn);
            float rs = p0 + p1;
            #pragma unroll
            for (int off = 1; off < 16; off <<= 1)
                rs += __shfl_xor(rs, off, 32);
            ls[r] = ls[r] * alpha + rs;
            m[r]  = mn;
            o0[r] *= alpha; o1[r] *= alpha; o2[r] *= alpha; o3[r] *= alpha;

            // C-layout -> LDS, to rebuild as an A fragment (wave-private buffer;
            // DS ops are in-order within a wave, so no barrier needed).
            const int prow = lo ? r : 8 + r;
            Ps[wave][prow * 32 + rlo]      = (__bf16)p0;
            Ps[wave][prow * 32 + 16 + rlo] = (__bf16)p1;
        }
        __asm__ volatile("" ::: "memory");

        FragBF pa;
        pa.q[0] = *(const u32x4v*)&Ps[wave][rlo * 32 + kB];
        pa.q[1] = *(const u32x4v*)&Ps[wave][rlo * 32 + kB + 8];

        // O += P(16x32) * V(32x64), four 16-wide e tiles.
        #pragma unroll
        for (int j = 0; j < 4; ++j) {
            FragBF vb;
            #pragma unroll
            for (int i = 0; i < 16; ++i)
                vb.e[i] = Vs[(kB + i) * 64 + j * 16 + rlo];
            v8f& o = (j == 0) ? o0 : (j == 1) ? o1 : (j == 2) ? o2 : o3;
            o = wmma_bf16(pa.v, vb.v, o);
        }
    }

    // Normalize and store heads: [b*512 + q][h*64 + e] (bf16).
    #pragma unroll
    for (int r = 0; r < 8; ++r) {
        const float inv = 1.0f / ls[r];
        const int qrow  = q0 + wave * 16 + (lo ? r : 8 + r);
        const size_t base = ((size_t)b * 512 + qrow) * 512 + (size_t)h * 64;
        Heads[base + 0  + rlo] = (__bf16)(o0[r] * inv);
        Heads[base + 16 + rlo] = (__bf16)(o1[r] * inv);
        Heads[base + 32 + rlo] = (__bf16)(o2[r] * inv);
        Heads[base + 48 + rlo] = (__bf16)(o3[r] * inv);
    }
}

// ---------------------------------------------------------------------------
// Orchestration.  Inputs (f32 unless noted):
//   0:q(32,512,512) 1:k 2:v 3:mask(32,512 bool) 4:W_query(8,512,64)
//   5:W_key 6:W_val 7:W_out(8,64,512).  Output: (32,512,512) f32.
// Workspace: Qh/Kh/Vh (h, b*q, e) bf16 + Heads (b*q, h*e) bf16 = 64 MB.
// ---------------------------------------------------------------------------
extern "C" void kernel_launch(void* const* d_in, const int* in_sizes, int n_in,
                              void* d_out, int out_size, void* d_ws, size_t ws_size,
                              hipStream_t stream)
{
    (void)in_sizes; (void)n_in; (void)out_size; (void)ws_size;

    const float* q  = (const float*)d_in[0];
    const float* k  = (const float*)d_in[1];
    const float* v  = (const float*)d_in[2];
    const unsigned char* mask = (const unsigned char*)d_in[3];
    const float* Wq = (const float*)d_in[4];
    const float* Wk = (const float*)d_in[5];
    const float* Wv = (const float*)d_in[6];
    const float* Wo = (const float*)d_in[7];
    float* out = (float*)d_out;

    char* ws = (char*)d_ws;
    const size_t SZ = (size_t)8 * 32 * 512 * 64 * sizeof(__bf16);  // 16 MB per tensor
    __bf16* Qh    = (__bf16*)(ws + 0 * SZ);
    __bf16* Kh    = (__bf16*)(ws + 1 * SZ);
    __bf16* Vh    = (__bf16*)(ws + 2 * SZ);
    __bf16* Heads = (__bf16*)(ws + 3 * SZ);

    const dim3 blk(128);

    // Per-head projections: [16384 x 512] x [512 x (8x64)] -> bf16 (h, bq, e).
    // HEADED grid: blockIdx.y covers all 8 heads -> A is streamed ONCE.
    const long wStride = 512L * 64;       // per-head weight stride (f32 elements)
    const long cStride = 16384L * 64;     // per-head output stride (bf16 elements)
    gemm_wmma_bf16<true,  true,  true ><<<dim3(1024, 8, 1), blk, 0, stream>>>(
        q, 512, Wq, 64, Qh, 64, 512, wStride, cStride);
    gemm_wmma_bf16<true,  true,  true ><<<dim3(1024, 8, 1), blk, 0, stream>>>(
        k, 512, Wk, 64, Kh, 64, 512, wStride, cStride);
    gemm_wmma_bf16<true,  true,  true ><<<dim3(1024, 8, 1), blk, 0, stream>>>(
        v, 512, Wv, 64, Vh, 64, 512, wStride, cStride);

    // Attention: grid (NQ/64, B, H).
    attn_flash_wmma<<<dim3(8, 32, 8), blk, 0, stream>>>(Qh, Kh, Vh, mask, Heads);

    // Output projection: [16384 x 512] x [512 x 512] -> f32.
    gemm_wmma_bf16<false, false, false><<<dim3(1024, 8, 1), blk, 0, stream>>>(
        Heads, 512, Wo, 512, out, 512, 512, 0, 0);
}
